// CausalSelfAttention_30863634989725
// MI455X (gfx1250) — matrix operand order repaired and checked by reading, not compiled, not run on hardware
//
#include <hip/hip_runtime.h>
#include <hip/hip_bf16.h>

// CDNA5 / gfx1250 causal self-attention.
//   f32->f16 convert
//   -> WMMA QKV projection: W block async-staged to LDS once (8x traffic cut),
//      V stored transposed
//   -> flash-attention: 256-wide output slices, async double-buffered K/V in LDS,
//      WMMA for Q@K^T and P@V, online softmax.

typedef _Float16 f16;
typedef _Float16 v16h __attribute__((ext_vector_type(16)));
typedef _Float16 v8h  __attribute__((ext_vector_type(8)));
typedef float    v8f  __attribute__((ext_vector_type(8)));

#define B_ 4
#define T_ 2048
#define C_ 1024
#define M_ (B_ * T_)          // 8192 tokens total

__device__ __forceinline__ v8f wmma_f16(v16h a, v16h b, v8f c) {
    return __builtin_amdgcn_wmma_f32_16x16x32_f16(false, a, false, b,
                                                  (short)0, c, false, false);
}

// Async global->LDS copy, 16B per lane, tracked by ASYNCcnt (gfx1250).
__device__ __forceinline__ void async_b128(unsigned lds_off, const void* gaddr) {
    asm volatile("global_load_async_to_lds_b128 %0, %1, off"
                 :: "v"(lds_off), "v"(gaddr) : "memory");
}
__device__ __forceinline__ void wait_async0() {
    asm volatile("s_wait_asynccnt 0x0" ::: "memory");
}
// Generic shared pointer -> 32-bit LDS byte offset (flat/async LDS addr = addr[31:0]).
__device__ __forceinline__ unsigned lds_off(const void* p) {
    return (unsigned)(unsigned long long)p;
}

// ---------------------------------------------------------------------------
// f32 -> f16 conversion
// ---------------------------------------------------------------------------
__global__ void k_f32_to_f16(const float* __restrict__ src,
                             f16* __restrict__ dst, int n) {
    int i = blockIdx.x * blockDim.x + threadIdx.x;
    if (i < n) dst[i] = (f16)src[i];
}

// ---------------------------------------------------------------------------
// QKV projection: out[m] = x @ W[m]^T + b[m]
//   grid = (C/64, M/128, 3), block = 256 (8 waves)
//   W block (64 out-cols x full K = contiguous 128KB) async-staged to LDS once.
//   m==0 -> qh (scaled by 1/sqrt(C)), m==1 -> kh, m==2 -> vt (transposed [b][c][t])
// ---------------------------------------------------------------------------
__global__ __launch_bounds__(256)
void k_qkv_proj(const f16* __restrict__ xh, const f16* __restrict__ wh,
                const float* __restrict__ bq, const float* __restrict__ bk,
                const float* __restrict__ bv,
                f16* __restrict__ qh, f16* __restrict__ kh, f16* __restrict__ vt)
{
    __shared__ __align__(32) f16 sW[64 * C_];       // 128KB: [n 0..63][k]

    const int tid  = threadIdx.x;
    const int lane = tid & 31;
    const int wave = tid >> 5;
    const int hi   = lane >> 4;
    const int ln   = lane & 15;
    const int m    = blockIdx.z;      // 0=Q 1=K 2=V
    const int rowBase = blockIdx.y * 128 + wave * 16;
    const int colBase = blockIdx.x * 64;
    const f16* wp = wh + (size_t)m * C_ * C_;

    // stage W[colBase..colBase+63][0..C) -- a contiguous 128KB block -> LDS
    {
        const char* g = (const char*)(wp + (size_t)colBase * C_) + tid * 512;
        unsigned     l = lds_off(&sW[0]) + (unsigned)(tid * 512);
#pragma unroll
        for (int q = 0; q < 32; ++q)
            async_b128(l + q * 16, g + q * 16);
        wait_async0();
        __syncthreads();
    }

    v8f acc[4] = {};

    const f16* arow = xh + (size_t)(rowBase + ln) * C_;
    for (int k0 = 0; k0 < C_; k0 += 32) {
        union { v16h v; v8h h[2]; } a;
        a.h[0] = *(const v8h*)(arow + k0 + hi * 16);
        a.h[1] = *(const v8h*)(arow + k0 + hi * 16 + 8);
#pragma unroll
        for (int cg = 0; cg < 4; ++cg) {
            // B fragment: lane -> local row cg*16+ln, K = k0 + hi*16 .. +15
            v16h bfrag = *(const v16h*)&sW[(cg * 16 + ln) * C_ + k0 + hi * 16];
            acc[cg] = wmma_f16(a.v, bfrag, acc[cg]);
        }
    }

    const float qscale = 0.03125f;    // 1/sqrt(1024), folded into Q
#pragma unroll
    for (int cg = 0; cg < 4; ++cg) {
        const int col = colBase + cg * 16 + ln;
        const float bias = (m == 0) ? bq[col] : (m == 1) ? bk[col] : bv[col];
        if (m == 2) {
            const int row0 = rowBase + 8 * hi;    // tiles never cross batch (128 | 2048)
            const int bb = row0 / T_;
            const int t0 = row0 % T_;
            v8h pack;
#pragma unroll
            for (int r = 0; r < 8; ++r) pack[r] = (f16)(acc[cg][r] + bias);
            *(v8h*)(vt + ((size_t)bb * C_ + col) * T_ + t0) = pack;
        } else {
            f16* dst = (m == 0) ? qh : kh;
            const float s = (m == 0) ? qscale : 1.0f;
#pragma unroll
            for (int r = 0; r < 8; ++r) {
                const int row = rowBase + r + 8 * hi;
                dst[(size_t)row * C_ + col] = (f16)((acc[cg][r] + bias) * s);
            }
        }
    }
}

// ---------------------------------------------------------------------------
// Flash attention with online softmax + async double-buffered K/V LDS staging.
//   grid = (C/256, T/128, B), block = 256 (8 waves)
//   wave: 16 query rows x 256 output cols; 32 keys per step.
//   LDS: sK 2x64KB, sV 2x16KB, sP 8KB  (168KB of 320KB/WGP)
// ---------------------------------------------------------------------------
__global__ __launch_bounds__(256)
void k_flash_attn(const f16* __restrict__ qh, const f16* __restrict__ kh,
                  const f16* __restrict__ vt, float* __restrict__ out)
{
    __shared__ __align__(32) f16 sK[2][32 * C_];    // [buf][key 0..31][c]
    __shared__ __align__(32) f16 sV[2][256 * 32];   // [buf][col 0..255][t 0..31]
    __shared__ __align__(32) f16 sP[8][16 * 32];    // per-wave P transit

    const int tid  = threadIdx.x;
    const int lane = tid & 31;
    const int wave = tid >> 5;
    const int hi   = lane >> 4;
    const int ln   = lane & 15;
    const int b    = blockIdx.z;
    const int rowBase128 = blockIdx.y * 128;
    const int qBase   = rowBase128 + wave * 16;
    const int colBase = blockIdx.x * 256;
    const int jEndWG   = rowBase128 + 128;   // workgroup-uniform causal bound
    const int jEndWave = qBase + 16;         // this wave's causal bound

    const f16* qrow  = qh + (size_t)(b * T_ + qBase + ln) * C_;
    const f16* kbase = kh + (size_t)b * T_ * C_;
    const f16* vbase = vt + (size_t)b * C_ * T_;

    // cooperative async staging of one 32-key tile into buffer `bf`
    auto stage_tile = [&](int bf, int j) {
        // K tile: rows j..j+31 are a contiguous 64KB block of kh -> linear copy.
        const char* gK = (const char*)(kbase + (size_t)j * C_) + tid * 256;
        unsigned     lK = lds_off(&sK[bf][0]) + (unsigned)(tid * 256);
#pragma unroll
        for (int q = 0; q < 16; ++q)
            async_b128(lK + q * 16, gK + q * 16);
        // V tile: 64B per column from vt (pre-transposed); 1 col (4 x b128) per thread.
        const char* gV = (const char*)(vbase + (size_t)(colBase + tid) * T_ + j);
        unsigned     lV = lds_off(&sV[bf][0]) + (unsigned)(tid * 64);
#pragma unroll
        for (int q = 0; q < 4; ++q)
            async_b128(lV + q * 16, gV + q * 16);
    };

    v8f o[16] = {};
    float mrow[8], lrow[8];
#pragma unroll
    for (int r = 0; r < 8; ++r) { mrow[r] = -1e30f; lrow[r] = 0.0f; }

    stage_tile(0, 0);
    wait_async0();
    __syncthreads();

    int buf = 0;
    for (int j = 0; j < jEndWG; j += 32) {
        if (j + 32 < jEndWG) stage_tile(buf ^ 1, j + 32);   // prefetch next tile

        if (j < jEndWave) {
            // ---- S = Qscaled @ K^T (two 16x16 tiles over 32 keys) ----
            v8f s0 = {}, s1 = {};
            const f16* k0p = &sK[buf][(size_t)ln * C_ + hi * 16];
            const f16* k1p = &sK[buf][(size_t)(16 + ln) * C_ + hi * 16];
#pragma unroll 4
            for (int kk = 0; kk < C_; kk += 32) {
                union { v16h v; v8h h[2]; } a;
                a.h[0] = *(const v8h*)(qrow + kk + hi * 16);
                a.h[1] = *(const v8h*)(qrow + kk + hi * 16 + 8);
                v16h b0 = *(const v16h*)(k0p + kk);
                v16h b1 = *(const v16h*)(k1p + kk);
                s0 = wmma_f16(a.v, b0, s0);
                s1 = wmma_f16(a.v, b1, s1);
            }

            // ---- causal mask + online softmax (C layout: M=r+8*hi, N=ln) ----
            const int kc0 = j + ln;
            const int kc1 = j + 16 + ln;
#pragma unroll
            for (int r = 0; r < 8; ++r) {
                const int qr = qBase + r + 8 * hi;
                float x0 = (kc0 <= qr) ? s0[r] : -1e30f;
                float x1 = (kc1 <= qr) ? s1[r] : -1e30f;
                float v = fmaxf(x0, x1);
#pragma unroll
                for (int msk = 8; msk >= 1; msk >>= 1)
                    v = fmaxf(v, __shfl_xor(v, msk, 32));
                const float mnew  = fmaxf(mrow[r], v);
                const float alpha = __expf(mrow[r] - mnew);
                const float p0 = __expf(x0 - mnew);
                const float p1 = __expf(x1 - mnew);
                float rs = p0 + p1;
#pragma unroll
                for (int msk = 8; msk >= 1; msk >>= 1)
                    rs += __shfl_xor(rs, msk, 32);
                lrow[r] = lrow[r] * alpha + rs;
                mrow[r] = mnew;
#pragma unroll
                for (int cg = 0; cg < 16; ++cg) o[cg][r] *= alpha;
                const int mloc = r + 8 * hi;
                sP[wave][mloc * 32 + ln]      = (f16)p0;
                sP[wave][mloc * 32 + 16 + ln] = (f16)p1;
            }

            asm volatile("s_wait_dscnt 0x0" ::: "memory");  // wave-local LDS RAW fence

            // ---- reload P as A fragment (16x32) ----
            union { v16h v; v8h h[2]; } ap;
            ap.h[0] = *(const v8h*)&sP[wave][ln * 32 + hi * 16];
            ap.h[1] = *(const v8h*)&sP[wave][ln * 32 + hi * 16 + 8];

            // ---- O += P @ V from staged V^T tile (256 cols) ----
#pragma unroll
            for (int cg = 0; cg < 16; ++cg) {
                const f16* vp = &sV[buf][(cg * 16 + ln) * 32 + hi * 16];
                v16h bvf = *(const v16h*)vp;
                o[cg] = wmma_f16(ap.v, bvf, o[cg]);
            }
        }

        wait_async0();       // next buffer landed
        __syncthreads();     // everyone done with current buffer
        buf ^= 1;
    }

    // ---- normalize and store f32 output ----
#pragma unroll
    for (int r = 0; r < 8; ++r) {
        const float inv = 1.0f / lrow[r];
        const int qr = qBase + r + 8 * hi;
        float* orow = out + (size_t)(b * T_ + qr) * C_ + colBase + ln;
#pragma unroll
        for (int cg = 0; cg < 16; ++cg) orow[cg * 16] = o[cg][r] * inv;
    }
}

// ---------------------------------------------------------------------------
// Host launcher
// ---------------------------------------------------------------------------
extern "C" void kernel_launch(void* const* d_in, const int* in_sizes, int n_in,
                              void* d_out, int out_size, void* d_ws, size_t ws_size,
                              hipStream_t stream)
{
    (void)in_sizes; (void)n_in; (void)out_size; (void)ws_size;
    const float* x  = (const float*)d_in[0];
    const float* Wq = (const float*)d_in[1];
    const float* bq = (const float*)d_in[2];
    const float* Wk = (const float*)d_in[3];
    const float* bk = (const float*)d_in[4];
    const float* Wv = (const float*)d_in[5];
    const float* bv = (const float*)d_in[6];
    float* out = (float*)d_out;

    // workspace: xh | Wqkv(f16) | qh | kh | vt  (~73 MB)
    f16* xh = (f16*)d_ws;
    f16* wh = xh + (size_t)M_ * C_;
    f16* qh = wh + (size_t)3 * C_ * C_;
    f16* kh = qh + (size_t)M_ * C_;
    f16* vt = kh + (size_t)M_ * C_;

    const int n  = M_ * C_;
    const int nw = C_ * C_;
    k_f32_to_f16<<<(n  + 255) / 256, 256, 0, stream>>>(x,  xh, n);
    k_f32_to_f16<<<(nw + 255) / 256, 256, 0, stream>>>(Wq, wh,          nw);
    k_f32_to_f16<<<(nw + 255) / 256, 256, 0, stream>>>(Wk, wh + nw,     nw);
    k_f32_to_f16<<<(nw + 255) / 256, 256, 0, stream>>>(Wv, wh + 2 * nw, nw);

    dim3 gProj(C_ / 64, M_ / 128, 3);
    k_qkv_proj<<<gProj, 256, 0, stream>>>(xh, wh, bq, bk, bv, qh, kh, vt);

    dim3 gAttn(C_ / 256, T_ / 128, B_);
    k_flash_attn<<<gAttn, 256, 0, stream>>>(qh, kh, vt, out);
}
